// ClusterMemory_42528766165029
// MI455X (gfx1250) — compile-verified, matrix operand order checked by reading, not run on hardware
//
#include <hip/hip_runtime.h>
#include <hip/hip_fp16.h>
#include <math.h>

typedef __attribute__((ext_vector_type(16))) _Float16 v16h;
typedef __attribute__((ext_vector_type(8)))  _Float16 v8h;
typedef __attribute__((ext_vector_type(4)))  _Float16 v4h;
typedef __attribute__((ext_vector_type(8)))  float    v8f;

#define B_ROWS   4096
#define N_COLS   131072
#define D_K      256
#define INV_TEMP 20.0f
#define EXP2_SCALE 28.853900817779268f     // INV_TEMP * log2(e): exp(20x)=2^(28.85x)
#define MTILE    128                       // rows per block (8 waves x 16)
#define NSPLIT   32                        // N-chunks across blockIdx.y
#define NTILES   (N_COLS / NSPLIT / 16)    // 256 16-col tiles per block
#define BPAD     8                         // halves of row padding (bank spread)

// ---------------------------------------------------------------------------
// CDNA5 async global->LDS copy (ASYNCcnt-tracked), ISA 15.18.3 op 98.
// The LDS *pointer* is passed into the asm (ptrtoint -> address escapes), so
// the "memory" clobber provably covers the shared tile: the compiler must keep
// the tile's ds_loads live instead of folding them to undef. Per ISA 10.2, a
// generic LDS address carries the LDS byte offset in addr[31:0], which is the
// VDST operand global_load_async_to_lds expects.
// ---------------------------------------------------------------------------
__device__ __forceinline__ void async_b128_to_lds(void* lds_ptr, const void* gptr) {
    unsigned           ldsa = (unsigned)(uintptr_t)lds_ptr;       // addr[31:0]
    unsigned long long ga   = (unsigned long long)(uintptr_t)gptr;
    asm volatile("global_load_async_to_lds_b128 %0, %1, off"
                 :: "v"(ldsa), "v"(ga) : "memory");
}
__device__ __forceinline__ void wait_async_all() {
    asm volatile("s_wait_asynccnt 0x0" ::: "memory");
}

// ---------------------------------------------------------------------------
// Kernel 0: features f32 -> f16 (done once; 64 MB stays L2-resident).
// ---------------------------------------------------------------------------
__global__ __launch_bounds__(256)
void cm_convert_feats(const float* __restrict__ f, _Float16* __restrict__ fh) {
    const size_t i = ((size_t)blockIdx.x * 256 + threadIdx.x) * 4;
    const float4 v = *(const float4*)(f + i);
    v4h o; o[0] = (_Float16)v.x; o[1] = (_Float16)v.y;
           o[2] = (_Float16)v.z; o[3] = (_Float16)v.w;
    *(v4h*)(fh + i) = o;
}

// ---------------------------------------------------------------------------
// Kernel 1: L2-normalize input rows, store f16.
// ---------------------------------------------------------------------------
__global__ __launch_bounds__(256)
void cm_normalize(const float* __restrict__ in, _Float16* __restrict__ xh) {
    __shared__ float red[256];
    const int row = blockIdx.x, tid = threadIdx.x;
    float v = in[(size_t)row * D_K + tid];
    red[tid] = v * v;
    __syncthreads();
    for (int s = 128; s > 0; s >>= 1) {
        if (tid < s) red[tid] += red[tid + s];
        __syncthreads();
    }
    const float scale = 1.0f / fmaxf(sqrtf(red[0]), 1e-12f);
    xh[(size_t)row * D_K + tid] = (_Float16)(v * scale);
}

// ---------------------------------------------------------------------------
// Kernel 2: WMMA GEMM + fused exp-sum. Block = 128 rows x (N/32) cols.
// B tiles (16x256 f16) are async-copied into double-buffered LDS and shared
// by all 8 waves; each wave keeps its 16-row A tile hoisted in 64 VGPRs.
// Logits are bounded (|dot|<=~1, /0.05 -> +-20.5), so a plain f32 exp-sum is
// safe: no online-max bookkeeping, one v_exp per logit on the TRANS pipe.
// ---------------------------------------------------------------------------
__global__ __launch_bounds__(256)
void cm_gemm_softmax(const _Float16* __restrict__ xh,
                     const _Float16* __restrict__ fh,
                     float* __restrict__ psum) {
    __shared__ __attribute__((aligned(16))) _Float16 bt[2][16][D_K + BPAD];

    const int tid  = threadIdx.x;
    const int wave = tid >> 5, lane = tid & 31;
    const int grp  = lane >> 4, ln = lane & 15;
    const int mbase  = blockIdx.x * MTILE + wave * 16;
    const int nstart = blockIdx.y * (N_COLS / NSPLIT);

    // Hoist this wave's 16x256 A tile: ISA 7.12.2 layout -> per lane, per K-slab,
    // halves 0-7 = K[8g..8g+7], halves 8-15 = K[16+8g..23+8g] (two b128 loads).
    v16h afrag[8];
    {
        const _Float16* xrow = xh + (size_t)(mbase + ln) * D_K;
#pragma unroll
        for (int kb = 0; kb < 8; ++kb) {
            v8h lo = *(const v8h*)(xrow + kb * 32 + 8 * grp);
            v8h hi = *(const v8h*)(xrow + kb * 32 + 16 + 8 * grp);
            afrag[kb] = __builtin_shufflevector(lo, hi,
                0,1,2,3,4,5,6,7,8,9,10,11,12,13,14,15);
        }
    }

    // B-tile loader mapping: thread t -> row t/16, 32B chunk (t%16)*16 halves.
    const int lr = tid >> 4;
    const int lc = (tid & 15) * 16;

    // Preload tile 0 into buffer 0.
    {
        const _Float16* src = fh + (size_t)(nstart + lr) * D_K + lc;
        _Float16*       dst = &bt[0][lr][lc];
        async_b128_to_lds(dst,     src);
        async_b128_to_lds(dst + 8, src + 8);
    }
    wait_async_all();
    __syncthreads();

    float rsum[8];
#pragma unroll
    for (int j = 0; j < 8; ++j) rsum[j] = 0.0f;

    for (int t = 0; t < NTILES; ++t) {
        const int buf = t & 1;
        if (t + 1 < NTILES) {   // async-prefetch next B tile into other buffer
            const _Float16* src = fh + (size_t)(nstart + (t + 1) * 16 + lr) * D_K + lc;
            _Float16*       dst = &bt[buf ^ 1][lr][lc];
            async_b128_to_lds(dst,     src);
            async_b128_to_lds(dst + 8, src + 8);
        }

        v8f c = {0.f,0.f,0.f,0.f,0.f,0.f,0.f,0.f};
        const _Float16* brow = &bt[buf][ln][0];
#pragma unroll
        for (int kb = 0; kb < 8; ++kb) {        // K = 256 in 8 chained WMMAs
            v8h lo = *(const v8h*)(brow + kb * 32 + 8 * grp);
            v8h hi = *(const v8h*)(brow + kb * 32 + 16 + 8 * grp);
            v16h b = __builtin_shufflevector(lo, hi,
                0,1,2,3,4,5,6,7,8,9,10,11,12,13,14,15);
            c = __builtin_amdgcn_wmma_f32_16x16x32_f16(
                    false, afrag[kb], false, b, (short)0, c, false, false);
        }

        // C layout: VGPR j = row (grp picks 0-7 / 8-15), lane%16 = column.
        // exp(logit) = 2^(c * 20*log2(e)) -> one v_mul + one v_exp per value.
#pragma unroll
        for (int j = 0; j < 8; ++j)
            rsum[j] += exp2f(c[j] * EXP2_SCALE);

        wait_async_all();      // my async writes for tile t+1 landed
        __syncthreads();       // everyone's landed; all reads of buf done
    }

    // Fold the 16 columns held across this half-wave's lanes.
#pragma unroll
    for (int j = 0; j < 8; ++j) {
        float s = rsum[j];
#pragma unroll
        for (int off = 8; off >= 1; off >>= 1)
            s += __shfl_xor(s, off, 16);
        if (ln == 0)
            psum[(size_t)(mbase + grp * 8 + j) * NSPLIT + blockIdx.y] = s;
    }
}

// ---------------------------------------------------------------------------
// Kernel 3: per-row finalize: lse = log(sum partials); target logit dot.
// ---------------------------------------------------------------------------
__global__ __launch_bounds__(256)
void cm_finalize_rows(const _Float16* __restrict__ xh,
                      const float*    __restrict__ feats,
                      const int*      __restrict__ targets,
                      const float*    __restrict__ psum,
                      float* __restrict__ rowloss) {
    const int wave = threadIdx.x >> 5, lane = threadIdx.x & 31;
    const int row  = blockIdx.x * 8 + wave;
    const int tgt  = targets[row];
    const float*    fr = feats + (size_t)tgt * D_K;
    const _Float16* xr = xh + (size_t)row * D_K;
    float acc = 0.0f;
    for (int k = lane; k < D_K; k += 32) acc += (float)xr[k] * fr[k];
#pragma unroll
    for (int off = 16; off >= 1; off >>= 1) acc += __shfl_xor(acc, off, 32);
    if (lane == 0) {
        float s = 0.0f;
        const float* ps = psum + (size_t)row * NSPLIT;
        for (int c = 0; c < NSPLIT; ++c) s += ps[c];
        rowloss[row] = __logf(s) - acc * INV_TEMP;
    }
}

// ---------------------------------------------------------------------------
// Kernel 4: mean over rows -> scalar loss.
// ---------------------------------------------------------------------------
__global__ __launch_bounds__(256)
void cm_reduce_loss(const float* __restrict__ rowloss, float* __restrict__ out) {
    __shared__ float red[256];
    float acc = 0.0f;
    for (int i = threadIdx.x; i < B_ROWS; i += 256) acc += rowloss[i];
    red[threadIdx.x] = acc;
    __syncthreads();
    for (int s = 128; s > 0; s >>= 1) {
        if (threadIdx.x < s) red[threadIdx.x] += red[threadIdx.x + s];
        __syncthreads();
    }
    if (threadIdx.x == 0) out[0] = red[0] / (float)B_ROWS;
}

// ---------------------------------------------------------------------------
extern "C" void kernel_launch(void* const* d_in, const int* in_sizes, int n_in,
                              void* d_out, int out_size, void* d_ws, size_t ws_size,
                              hipStream_t stream) {
    const float* inputs  = (const float*)d_in[0];   // [B, D] f32
    const int*   targets = (const int*)  d_in[1];   // [B] int
    const float* feats   = (const float*)d_in[4];   // [N, D] f32
    float*       out     = (float*)d_out;

    char*  ws  = (char*)d_ws;
    size_t off = 0;
    _Float16* fh = (_Float16*)(ws + off);                       // 64 MB
    off += (size_t)N_COLS * D_K * sizeof(_Float16);
    off = (off + 255) & ~(size_t)255;
    _Float16* xh = (_Float16*)(ws + off);                       // 2 MB
    off += (size_t)B_ROWS * D_K * sizeof(_Float16);
    off = (off + 255) & ~(size_t)255;
    float* psum = (float*)(ws + off);                           // 512 KB
    off += (size_t)B_ROWS * NSPLIT * sizeof(float);
    float* rowloss = (float*)(ws + off);                        // 16 KB

    cm_convert_feats <<<(N_COLS * D_K / 4) / 256, 256, 0, stream>>>(feats, fh);
    cm_normalize     <<<B_ROWS, 256, 0, stream>>>(inputs, xh);
    cm_gemm_softmax  <<<dim3(B_ROWS / MTILE, NSPLIT), 256, 0, stream>>>(xh, fh, psum);
    cm_finalize_rows <<<B_ROWS / 8, 256, 0, stream>>>(xh, feats, targets, psum, rowloss);
    cm_reduce_loss   <<<1, 256, 0, stream>>>(rowloss, out);
}
